// LSTM_Model_42331197669775
// MI455X (gfx1250) — matrix operand order repaired and checked by reading, not compiled, not run on hardware
//
#include <hip/hip_runtime.h>
#include <hip/hip_bf16.h>

// ---------------- problem constants ----------------
#define BB 256   // batch
#define TT 256   // timesteps
#define II 512   // input size
#define HH 512   // hidden size
#define N4 2048  // 4*H gate width
#define KK 1024  // I + H (fused GEMM K)

typedef __attribute__((ext_vector_type(16))) __bf16 v16bf;
typedef __attribute__((ext_vector_type(8)))  __bf16 v8bf;
typedef __attribute__((ext_vector_type(8)))  float  v8f;
typedef __attribute__((ext_vector_type(4)))  float  v4f;

__device__ __forceinline__ float fsigmoid(float x) {
    return 1.0f / (1.0f + __expf(-x));
}
__device__ __forceinline__ float ftanh(float x) {
    // tanh(x) = 1 - 2/(1+exp(2x)); saturates correctly at +/-inf
    return 1.0f - 2.0f / (1.0f + __expf(2.0f * x));
}

// ---------------- prep: transpose+convert weights, fold biases, zero state --------
// Wt[k][n] (bf16, k-major, 1024x2048): k<512 -> W_ih[n][k], else W_hh[n][k-512]
__global__ __launch_bounds__(256)
void lstm_prep(const float* __restrict__ W_ih, const float* __restrict__ W_hh,
               const float* __restrict__ b_ih, const float* __restrict__ b_hh,
               __bf16* __restrict__ Wt, float* __restrict__ bias,
               __bf16* __restrict__ hA, float* __restrict__ c)
{
    int idx = blockIdx.x * blockDim.x + threadIdx.x;
    if (idx < KK * N4) {
        int k = idx >> 11;          // / 2048
        int n = idx & (N4 - 1);     // % 2048
        float v = (k < II) ? W_ih[(size_t)n * II + k]
                           : W_hh[(size_t)n * HH + (k - II)];
        Wt[idx] = (__bf16)v;
    }
    if (idx < N4) bias[idx] = b_ih[idx] + b_hh[idx];
    if (idx < BB * HH) { hA[idx] = (__bf16)0.0f; c[idx] = 0.0f; }
}

// ---------------- one LSTM timestep: split-K dual GEMM + gate nonlinearity -------
// grid = (B/16, H/64), block = 256 (8 waves = 4 wave-pairs).
// Wave-pair p owns output tile (mbase, nbase = by*64 + p*16) for ALL four gates.
//   half 0: K in [0,512)    -> x @ W_ih^T   (f32 loads, cvt to bf16)
//   half 1: K in [512,1024) -> h @ W_hh^T   (bf16 loads)
// half-1 dumps 4 partial 16x16 f32 accumulators to LDS; half-0 reduces and runs
// the sigmoid/tanh/cell-update epilogue entirely in registers.
__global__ __launch_bounds__(256)
void lstm_step(const float*  __restrict__ x,      // [B,T,I] f32
               const __bf16* __restrict__ Wt,     // [1024][2048] bf16 k-major
               const float*  __restrict__ bias,   // [2048] f32 (b_ih+b_hh)
               const __bf16* __restrict__ hprev,  // [B,H] bf16
               __bf16*       __restrict__ hnext,  // [B,H] bf16
               float*        __restrict__ c,      // [B,H] f32 (in-place)
               int t)
{
    __shared__ float red[4][4][256];   // [pair][gate][lane*8] = 16 KB

    const int lane  = threadIdx.x & 31;
    const int wave  = threadIdx.x >> 5;
    const int pair  = wave >> 1;        // tile index within block (0..3)
    const int half  = wave & 1;         // 0 = x K-half, 1 = h K-half
    const int lq    = lane & 15;        // A row / D col
    const int lh    = lane >> 4;        // K-octet half / D row-block
    const int mbase = blockIdx.x * 16;
    const int nbase = blockIdx.y * 64 + pair * 16;
    const int m     = mbase + lq;

    v8f acc[4] = {v8f{}, v8f{}, v8f{}, v8f{}};

    if (half == 0) {
        // ---- K in [0,512): input contribution, f32 -> bf16 on the fly ----
        const float* xrow = x + ((size_t)m * TT + t) * II;
        for (int kb = 0; kb < II; kb += 32) {
            v4f x0 = *(const v4f*)(xrow + kb + 8 * lh);
            v4f x1 = *(const v4f*)(xrow + kb + 8 * lh + 4);
            v4f x2 = *(const v4f*)(xrow + kb + 16 + 8 * lh);
            v4f x3 = *(const v4f*)(xrow + kb + 16 + 8 * lh + 4);
            v16bf a;
#pragma unroll
            for (int j = 0; j < 4; ++j) {
                a[j]      = (__bf16)x0[j];
                a[4 + j]  = (__bf16)x1[j];
                a[8 + j]  = (__bf16)x2[j];
                a[12 + j] = (__bf16)x3[j];
            }
            const __bf16* wk = Wt + (size_t)kb * N4 + nbase + lq;
            v16bf b0, b1, b2, b3;
#pragma unroll
            for (int s = 0; s < 16; ++s) {
                const size_t o = (size_t)((s < 8) ? (8 * lh + s) : (16 + 8 * lh + s - 8)) * N4;
                b0[s] = wk[o];
                b1[s] = wk[o + HH];
                b2[s] = wk[o + 2 * HH];
                b3[s] = wk[o + 3 * HH];
            }
            acc[0] = __builtin_amdgcn_wmma_f32_16x16x32_bf16(false, a, false, b0, (short)0, acc[0], false, false);
            acc[1] = __builtin_amdgcn_wmma_f32_16x16x32_bf16(false, a, false, b1, (short)0, acc[1], false, false);
            acc[2] = __builtin_amdgcn_wmma_f32_16x16x32_bf16(false, a, false, b2, (short)0, acc[2], false, false);
            acc[3] = __builtin_amdgcn_wmma_f32_16x16x32_bf16(false, a, false, b3, (short)0, acc[3], false, false);
        }
    } else {
        // ---- K in [512,1024): recurrent contribution, bf16 direct loads ----
        const __bf16* hrow = hprev + (size_t)m * HH;
        for (int kb = 0; kb < HH; kb += 32) {
            v8bf h0 = *(const v8bf*)(hrow + kb + 8 * lh);
            v8bf h1 = *(const v8bf*)(hrow + kb + 16 + 8 * lh);
            v16bf a;
#pragma unroll
            for (int j = 0; j < 8; ++j) { a[j] = h0[j]; a[8 + j] = h1[j]; }
            const __bf16* wk = Wt + (size_t)(II + kb) * N4 + nbase + lq;
            v16bf b0, b1, b2, b3;
#pragma unroll
            for (int s = 0; s < 16; ++s) {
                const size_t o = (size_t)((s < 8) ? (8 * lh + s) : (16 + 8 * lh + s - 8)) * N4;
                b0[s] = wk[o];
                b1[s] = wk[o + HH];
                b2[s] = wk[o + 2 * HH];
                b3[s] = wk[o + 3 * HH];
            }
            acc[0] = __builtin_amdgcn_wmma_f32_16x16x32_bf16(false, a, false, b0, (short)0, acc[0], false, false);
            acc[1] = __builtin_amdgcn_wmma_f32_16x16x32_bf16(false, a, false, b1, (short)0, acc[1], false, false);
            acc[2] = __builtin_amdgcn_wmma_f32_16x16x32_bf16(false, a, false, b2, (short)0, acc[2], false, false);
            acc[3] = __builtin_amdgcn_wmma_f32_16x16x32_bf16(false, a, false, b3, (short)0, acc[3], false, false);
        }
        // dump partial sums for the x-half wave (ds_store_b128 x2 per gate)
#pragma unroll
        for (int g = 0; g < 4; ++g)
            *(v8f*)&red[pair][g][lane * 8] = acc[g];
    }

    __syncthreads();
    if (half == 1) return;

    // ---- reduce split-K partials ----
#pragma unroll
    for (int g = 0; g < 4; ++g) {
        v8f p = *(const v8f*)&red[pair][g][lane * 8];
        acc[g] = acc[g] + p;
    }

    // ---- epilogue: gates + cell update in registers ----
    const int ncol = nbase + lq;
    const float bi = bias[ncol];
    const float bf = bias[HH + ncol];
    const float bg = bias[2 * HH + ncol];
    const float bo = bias[3 * HH + ncol];
#pragma unroll
    for (int r = 0; r < 8; ++r) {
        const int    mr  = mbase + 8 * lh + r;     // D layout: VGPR r, lane-half lh
        const size_t idx = (size_t)mr * HH + ncol;
        float gi = fsigmoid(acc[0][r] + bi);
        float gf = fsigmoid(acc[1][r] + bf);
        float gg = ftanh   (acc[2][r] + bg);
        float go = fsigmoid(acc[3][r] + bo);
        float cn = gf * c[idx] + gi * gg;
        c[idx] = cn;
        hnext[idx] = (__bf16)(go * ftanh(cn));
    }
}

// ---------------- final linear head: out[b] = h_last[b,:] . fc_w + fc_b ----------
__global__ __launch_bounds__(256)
void lstm_head(const __bf16* __restrict__ h, const float* __restrict__ fc_w,
               const float* __restrict__ fc_b, float* __restrict__ out)
{
    int b = blockIdx.x * blockDim.x + threadIdx.x;
    if (b < BB) {
        float s = 0.0f;
        for (int k = 0; k < HH; ++k)
            s += (float)h[(size_t)b * HH + k] * fc_w[k];
        out[b] = s + fc_b[0];
    }
}

extern "C" void kernel_launch(void* const* d_in, const int* in_sizes, int n_in,
                              void* d_out, int out_size, void* d_ws, size_t ws_size,
                              hipStream_t stream) {
    (void)in_sizes; (void)n_in; (void)out_size; (void)ws_size;
    const float* x    = (const float*)d_in[0];
    const float* W_ih = (const float*)d_in[1];
    const float* W_hh = (const float*)d_in[2];
    const float* b_ih = (const float*)d_in[3];
    const float* b_hh = (const float*)d_in[4];
    const float* fc_w = (const float*)d_in[5];
    const float* fc_b = (const float*)d_in[6];

    // workspace layout (bytes, 256-aligned); total ~5.01 MB
    char* ws = (char*)d_ws;
    __bf16* Wt   = (__bf16*)(ws);                 // 1024*2048*2 = 4,194,304
    float*  bias = (float*) (ws + 4194304);       // 2048*4      =     8,192
    __bf16* hA   = (__bf16*)(ws + 4202496);       // 256*512*2   =   262,144
    __bf16* hB   = (__bf16*)(ws + 4464640);       // 256*512*2   =   262,144
    float*  cbuf = (float*) (ws + 4726784);       // 256*512*4   =   524,288

    // 1) weights -> bf16 k-major, fold biases, zero h0/c0 (re-done every call)
    lstm_prep<<<(KK * N4 + 255) / 256, 256, 0, stream>>>(
        W_ih, W_hh, b_ih, b_hh, Wt, bias, hA, cbuf);

    // 2) sequential recurrence: one kernel per timestep, double-buffered h,
    //    2-way split-K (x-half wave + h-half wave) per output tile.
    dim3 grid(BB / 16, HH / 64);    // 16 x 8 blocks, 8 waves each = 1024 waves
    for (int t = 0; t < TT; ++t) {
        const __bf16* hp = (t & 1) ? hB : hA;
        __bf16*       hn = (t & 1) ? hA : hB;
        lstm_step<<<grid, 256, 0, stream>>>(x, Wt, bias, hp, hn, cbuf, t);
    }
    // t=255 (odd) wrote hA -> h_last lives in hA

    // 3) head
    lstm_head<<<1, 256, 0, stream>>>(hA, fc_w, fc_b, (float*)d_out);
}
